// FPGA_GNN_GATv2_18511309046120
// MI455X (gfx1250) — compile-verified, matrix operand order checked by reading, not run on hardware
//
#include <hip/hip_runtime.h>
#include <hip/hip_bf16.h>
#include <math.h>

typedef __attribute__((ext_vector_type(16))) _Float16 v16h;
typedef __attribute__((ext_vector_type(8)))  _Float16 v8h;
typedef __attribute__((ext_vector_type(8)))  float    v8f;
typedef __attribute__((ext_vector_type(4)))  float    v4f;
typedef unsigned int u32x4 __attribute__((ext_vector_type(4)));
typedef int          i32x8 __attribute__((ext_vector_type(8)));
typedef int          i32x4 __attribute__((ext_vector_type(4)));

#define HEADS 4
#define CH    64
#define DDIM  256
#define SLOPE 0.2f
#define TILE_M 64

__device__ __forceinline__ float wave_sum(float v) {
#pragma unroll
  for (int o = 16; o > 0; o >>= 1) v += __shfl_xor(v, o, 32);
  return v;
}

__device__ __forceinline__ void atomic_max_f32(float* addr, float val) {
  if (val >= 0.f) atomicMax((int*)addr, __float_as_int(val));
  else            atomicMin((unsigned int*)addr, __float_as_uint(val));
}

// ---------------------------------------------------------------------------
// f32 -> f16 elementwise convert (activations)
// ---------------------------------------------------------------------------
__global__ void cvt_f32_f16(const float* __restrict__ in, _Float16* __restrict__ out,
                            int count)
{
  int i = blockIdx.x * blockDim.x + threadIdx.x;
  if (i < count) out[i] = (_Float16)in[i];
}

// f32 weight [K x ncols] -> f16 transposed Wt [ncols x K] (coalesced read)
__global__ void cvt_w_t(const float* __restrict__ W, _Float16* __restrict__ Wt,
                        int K, int ncols)
{
  int idx = blockIdx.x * blockDim.x + threadIdx.x;
  if (idx >= K * ncols) return;
  int k = idx / ncols, c = idx - k * ncols;
  Wt[(size_t)c * K + k] = (_Float16)W[idx];
}

// ---------------------------------------------------------------------------
// C[M x ncols] = Ah[M x K](f16) * Wt[ncols x K](f16, col-major) + bias.
// Block = (ncols/16 capped at 8) waves; block tile = 64 rows x 16*waves cols.
// A tile (64 x K f16) staged into LDS by the Tensor Data Mover with row
// padding (+16B) to avoid bank conflicts; each wave runs 4 WMMAs per K-step
// (4 stacked M-tiles) reusing one B fragment. A-fragments are read with
// explicit ds_load_b128 (inline asm) so the compiler cannot fold the reads
// of the TDM-written tile.
// ---------------------------------------------------------------------------
__global__ void __launch_bounds__(256)
gemm_f16wmma_tdm(const _Float16* __restrict__ Ah, const _Float16* __restrict__ Wt,
                 const float* __restrict__ bias, float* __restrict__ C,
                 int K, int ncols)
{
  __shared__ __attribute__((aligned(16))) _Float16 tileA[TILE_M * (DDIM + 8)];

  const int lane = threadIdx.x & 31;
  const int wave = threadIdx.x >> 5;
  const int nwaves = blockDim.x >> 5;
  const int m0 = blockIdx.x * TILE_M;
  const int n0 = (blockIdx.y * nwaves + wave) * 16;
  const int ldsStride = K + 8;               // halfs; +16B pad per row

  // --- TDM: stage A tile (TILE_M x K f16) global -> LDS, padded ------------
  if (wave == 0) {
    unsigned long long ga =
        (unsigned long long)(uintptr_t)(Ah + (size_t)m0 * K);
    // pad_interval code: one row = K/2 dwords = 2^(v+1)  -> v = log2(K/2)-1
    unsigned pint = (unsigned)(31 - __builtin_clz((unsigned)(K >> 1))) - 1u;
    unsigned pamt = 3u;                      // 4 dwords = 16B pad
    u32x4 g0;
    g0[0] = 1u;                                         // count=1, user mode
    g0[1] = 0u;                                         // lds_addr = 0 (tileA)
    g0[2] = (unsigned)(ga & 0xFFFFFFFFu);               // global_addr[31:0]
    g0[3] = (unsigned)((ga >> 32) & 0x1FFFFFFu) | (2u << 30);  // addr[56:32]|type=2
    i32x8 g1;
    g1[0] = (int)((1u << 16) | (1u << 20) | (pint << 22) | (pamt << 25));
    g1[1] = (int)(((unsigned)K & 0xFFFFu) << 16);       // tensor_dim0 lo
    g1[2] = (int)((((unsigned)K >> 16) & 0xFFFFu) | ((unsigned)TILE_M << 16));
    g1[3] = (int)(((unsigned)K & 0xFFFFu) << 16);       // tile_dim0 = K
    g1[4] = (int)((unsigned)TILE_M);                    // tile_dim1=64, tile_dim2=0
    g1[5] = (int)((unsigned)K);                         // dim0 stride lo
    g1[6] = 0;
    g1[7] = 0;
    i32x4 gz4 = {0, 0, 0, 0};
    i32x8 gz8 = {0, 0, 0, 0, 0, 0, 0, 0};
    __builtin_amdgcn_tensor_load_to_lds(g0, g1, gz4, gz4, gz8, 0);
    __builtin_amdgcn_s_wait_tensorcnt(0);
  }
  __syncthreads();

  const int mA = lane & 15;
  const int kg = (lane >> 4) * 8;
  const int nB = lane & 15;
  // LDS byte offset of tileA (low 32 bits of the generic address are the
  // wave-relative LDS address per the flat-aperture mapping)
  const unsigned tbase = (unsigned)(uintptr_t)tileA;

  v8f acc[4] = {};
  const _Float16* wrow = Wt + (size_t)(n0 + nB) * K + kg;

  for (int k0 = 0; k0 < K; k0 += 32) {
    v8h blo = *(const v8h*)(wrow + k0);
    v8h bhi = *(const v8h*)(wrow + k0 + 16);
    __builtin_prefetch(wrow + k0 + 32, 0, 0);
    v16h bf;
#pragma unroll
    for (int i = 0; i < 8; ++i) { bf[i] = blo[i]; bf[8 + i] = bhi[i]; }
#pragma unroll
    for (int t = 0; t < 4; ++t) {
      unsigned aoff =
          tbase + (unsigned)(((t * 16 + mA) * ldsStride + k0 + kg) * 2);
      v8h alo, ahi;
      // explicit LDS reads of the TDM-staged tile (+16 halfs = offset:32)
      asm volatile("ds_load_b128 %0, %2\n\t"
                   "ds_load_b128 %1, %2 offset:32\n\t"
                   "s_wait_dscnt 0"
                   : "=v"(alo), "=v"(ahi)
                   : "v"(aoff));
      v16h af;
#pragma unroll
      for (int i = 0; i < 8; ++i) { af[i] = alo[i]; af[8 + i] = ahi[i]; }
      acc[t] = __builtin_amdgcn_wmma_f32_16x16x32_f16(false, af, false, bf,
                                                      (short)0, acc[t],
                                                      false, false);
    }
  }

  // D layout: VGPR r -> M = r + 8*(lane>>4), N = lane&15
  const int n  = n0 + (lane & 15);
  const int mb = m0 + ((lane >> 4) ? 8 : 0);
  const float bn = bias[n];
#pragma unroll
  for (int t = 0; t < 4; ++t)
#pragma unroll
    for (int r = 0; r < 8; ++r)
      C[(size_t)(mb + t * 16 + r) * ncols + n] = acc[t][r] + bn;
}

// ---------------------------------------------------------------------------
__global__ void init_layer(float* __restrict__ acc, float* __restrict__ smax,
                           float* __restrict__ den, int n)
{
  int i = blockIdx.x * blockDim.x + threadIdx.x;
  if (i < n * DDIM) acc[i] = 0.f;
  if (i < n * HEADS) { smax[i] = -__builtin_inff(); den[i] = 0.f; }
}

// s[e,h] = att_h . leaky_relu(xl[src] + xr[dst]); fused segment-max over dst
__global__ void edge_score(const float* __restrict__ xl, const float* __restrict__ xr,
                           const float* __restrict__ att,
                           const int* __restrict__ src, const int* __restrict__ dst,
                           int E, int Etot, float* __restrict__ s,
                           float* __restrict__ smax)
{
  int idx = blockIdx.x * blockDim.x + threadIdx.x;
  if (idx >= Etot * HEADS) return;
  int e = idx >> 2, h = idx & 3;
  int sv = (e < E) ? src[e] : (e - E);   // self loops appended at the end
  int dv = (e < E) ? dst[e] : (e - E);
  const float* pl = xl + (size_t)sv * DDIM + h * CH;
  const float* pr = xr + (size_t)dv * DDIM + h * CH;
  const float* pa = att + h * CH;
  float acc = 0.f;
#pragma unroll 4
  for (int c = 0; c < CH; c += 4) {
    v4f l = *(const v4f*)(pl + c);
    v4f r = *(const v4f*)(pr + c);
    v4f a = *(const v4f*)(pa + c);
#pragma unroll
    for (int j = 0; j < 4; ++j) {
      float v = l[j] + r[j];
      v = (v > 0.f) ? v : SLOPE * v;
      acc += a[j] * v;
    }
  }
  s[idx] = acc;
  atomic_max_f32(&smax[dv * HEADS + h], acc);
}

// s[e,h] <- exp(s - smax[dst]); fused segment-sum into den
__global__ void edge_exp(float* __restrict__ s, const float* __restrict__ smax,
                         float* __restrict__ den,
                         const int* __restrict__ dst, int E, int Etot)
{
  int idx = blockIdx.x * blockDim.x + threadIdx.x;
  if (idx >= Etot * HEADS) return;
  int e = idx >> 2, h = idx & 3;
  int dv = (e < E) ? dst[e] : (e - E);
  float ex = __expf(s[idx] - smax[dv * HEADS + h]);
  s[idx] = ex;
  atomicAdd(&den[dv * HEADS + h], ex);
}

// acc[dst] += (ex/den[dst]) * xl[src]
__global__ void edge_scatter(const float* __restrict__ s, const float* __restrict__ den,
                             const float* __restrict__ xl,
                             const int* __restrict__ src, const int* __restrict__ dst,
                             int E, int Etot, float* __restrict__ acc)
{
  int idx = blockIdx.x * blockDim.x + threadIdx.x;
  if (idx >= Etot * HEADS) return;
  int e = idx >> 2, h = idx & 3;
  int sv = (e < E) ? src[e] : (e - E);
  int dv = (e < E) ? dst[e] : (e - E);
  float alpha = s[idx] / den[dv * HEADS + h];
  const float* pl = xl + (size_t)sv * DDIM + h * CH;
  float* po = acc + (size_t)dv * DDIM + h * CH;
#pragma unroll 4
  for (int c = 0; c < CH; ++c)
    atomicAdd(&po[c], alpha * pl[c]);
}

// h <- elu(layernorm(acc + gat_bias)) + idn   (in place on acc), wave per node
__global__ void __launch_bounds__(256)
node_post(float* __restrict__ acc, const float* __restrict__ gb,
          const float* __restrict__ g, const float* __restrict__ b,
          const float* __restrict__ idn, int n)
{
  int wave = threadIdx.x >> 5, lane = threadIdx.x & 31;
  int node = blockIdx.x * 8 + wave;
  if (node >= n) return;
  float* row = acc + (size_t)node * DDIM;
  const float* irow = idn + (size_t)node * DDIM;
  float v[8];
  float s = 0.f;
#pragma unroll
  for (int j = 0; j < 8; ++j) { int c = j * 32 + lane; v[j] = row[c] + gb[c]; s += v[j]; }
  s = wave_sum(s);
  float m = s * (1.f / 256.f);
  float vs = 0.f;
#pragma unroll
  for (int j = 0; j < 8; ++j) { float d = v[j] - m; vs += d * d; }
  vs = wave_sum(vs);
  float inv = rsqrtf(vs * (1.f / 256.f) + 1e-5f);
#pragma unroll
  for (int j = 0; j < 8; ++j) {
    int c = j * 32 + lane;
    float y = (v[j] - m) * inv * g[c] + b[c];
    y = (y > 0.f) ? y : (__expf(y) - 1.f);
    row[c] = y + irow[c];
  }
}

// per-graph sum / mean / max pooling with softmax(pool_w) mixing
__global__ void __launch_bounds__(256)
pool_kernel(const float* __restrict__ hf, const float* __restrict__ pool_w,
            float* __restrict__ emb, int P)
{
  __shared__ float ssum[256], smx[256];
  int gidx = blockIdx.x;
  int d = threadIdx.x & 63, part = threadIdx.x >> 6;
  float sum = 0.f, mx = -__builtin_inff();
  for (int i = part; i < P; i += 4) {
    float v = hf[(size_t)(gidx * P + i) * 64 + d];
    sum += v; mx = fmaxf(mx, v);
  }
  ssum[threadIdx.x] = sum; smx[threadIdx.x] = mx;
  __syncthreads();
  if (part == 0) {
    for (int p = 1; p < 4; ++p) {
      sum += ssum[p * 64 + d];
      mx = fmaxf(mx, smx[p * 64 + d]);
    }
    float w0 = __expf(pool_w[0]), w1 = __expf(pool_w[1]), w2 = __expf(pool_w[2]);
    float tw = 1.f / (w0 + w1 + w2);
    emb[gidx * 64 + d] = (w0 * sum + w1 * (sum / (float)P) + w2 * mx) * tw;
  }
}

// tiny MLP head: [emb|strategy](16x65) -> 64 -> 32 -> 6, LN+ReLU between.
__global__ void __launch_bounds__(512)
mlp_head(const float* __restrict__ emb, const float* __restrict__ strat,
         const float* __restrict__ W1, const float* __restrict__ b1,
         const float* __restrict__ g1, const float* __restrict__ bb1,
         const float* __restrict__ W2, const float* __restrict__ b2,
         const float* __restrict__ g2, const float* __restrict__ bb2,
         const float* __restrict__ W3, const float* __restrict__ b3,
         float* __restrict__ out)
{
  __shared__ float act1[16][64];
  __shared__ float act2[16][32];
  int gidx = threadIdx.x >> 5, lane = threadIdx.x & 31;

  float a0 = b1[lane], a1 = b1[lane + 32];
  for (int i = 0; i < 64; ++i) {
    float zi = emb[gidx * 64 + i];
    a0 += zi * W1[i * 64 + lane];
    a1 += zi * W1[i * 64 + lane + 32];
  }
  {
    float zi = strat[gidx];
    a0 += zi * W1[64 * 64 + lane];
    a1 += zi * W1[64 * 64 + lane + 32];
  }
  float m = wave_sum(a0 + a1) * (1.f / 64.f);
  float d0 = a0 - m, d1 = a1 - m;
  float inv = rsqrtf(wave_sum(d0 * d0 + d1 * d1) * (1.f / 64.f) + 1e-5f);
  a0 = fmaxf(d0 * inv * g1[lane] + bb1[lane], 0.f);
  a1 = fmaxf(d1 * inv * g1[lane + 32] + bb1[lane + 32], 0.f);
  act1[gidx][lane] = a0; act1[gidx][lane + 32] = a1;
  __syncthreads();

  float c = b2[lane];
  for (int i = 0; i < 64; ++i) c += act1[gidx][i] * W2[i * 32 + lane];
  m = wave_sum(c) * (1.f / 32.f);
  float dd = c - m;
  inv = rsqrtf(wave_sum(dd * dd) * (1.f / 32.f) + 1e-5f);
  c = fmaxf(dd * inv * g2[lane] + bb2[lane], 0.f);
  act2[gidx][lane] = c;
  __syncthreads();

  if (lane < 6) {
    float o = b3[lane];
    for (int i = 0; i < 32; ++i) o += act2[gidx][i] * W3[i * 6 + lane];
    out[gidx * 6 + lane] = o;
  }
}

// ---------------------------------------------------------------------------
// Host-side launch. Input order = setup_inputs() dict insertion order with
// params flattened recursively in insertion order (44 flat arrays).
// ---------------------------------------------------------------------------
extern "C" void kernel_launch(void* const* d_in, const int* in_sizes, int n_in,
                              void* d_out, int out_size, void* d_ws, size_t ws_size,
                              hipStream_t stream)
{
  const float* x        = (const float*)d_in[0];
  const float* strategy = (const float*)d_in[1];
  const float* Wl[3]  = { (const float*)d_in[2],  (const float*)d_in[8],  (const float*)d_in[14] };
  const float* bl[3]  = { (const float*)d_in[3],  (const float*)d_in[9],  (const float*)d_in[15] };
  const float* Wr[3]  = { (const float*)d_in[4],  (const float*)d_in[10], (const float*)d_in[16] };
  const float* br[3]  = { (const float*)d_in[5],  (const float*)d_in[11], (const float*)d_in[17] };
  const float* att[3] = { (const float*)d_in[6],  (const float*)d_in[12], (const float*)d_in[18] };
  const float* gb[3]  = { (const float*)d_in[7],  (const float*)d_in[13], (const float*)d_in[19] };
  const float* ln_g[3] = { (const float*)d_in[20], (const float*)d_in[21], (const float*)d_in[22] };
  const float* ln_b[3] = { (const float*)d_in[23], (const float*)d_in[24], (const float*)d_in[25] };
  const float* res_W = (const float*)d_in[26];
  const float* res_b = (const float*)d_in[27];
  const float* fp_W  = (const float*)d_in[28];
  const float* fp_b  = (const float*)d_in[29];
  const float* pool_w = (const float*)d_in[30];
  const float* W1 = (const float*)d_in[31]; const float* b1 = (const float*)d_in[32];
  const float* g1 = (const float*)d_in[33]; const float* bb1 = (const float*)d_in[34];
  const float* W2 = (const float*)d_in[35]; const float* b2 = (const float*)d_in[36];
  const float* g2 = (const float*)d_in[37]; const float* bb2 = (const float*)d_in[38];
  const float* W3 = (const float*)d_in[39]; const float* b3 = (const float*)d_in[40];
  const int* esrc = (const int*)d_in[41];
  const int* edst = (const int*)d_in[42];

  const int n    = in_sizes[0] / 64;   // 32000 nodes
  const int E    = in_sizes[41];       // 512000 edges
  const int Etot = E + n;              // + self loops
  const int G    = in_sizes[1];        // 16 graphs
  const int P    = n / G;

  // workspace carve-up (floats)
  float* ws = (float*)d_ws;
  const size_t NB = (size_t)n * DDIM;
  float* bufA  = ws;
  float* bufB  = ws + NB;
  float* xlp   = ws + 2 * NB;
  float* xrp   = ws + 3 * NB;
  float* sbuf  = ws + 4 * NB;
  float* smaxb = sbuf + (size_t)Etot * HEADS;
  float* denb  = smaxb + (size_t)n * HEADS;
  float* embb  = denb + (size_t)n * HEADS;
  _Float16* Ah  = (_Float16*)(embb + 1024);          // n*256 halfs
  _Float16* Wt0 = Ah + NB;                           // 256*256 halfs
  _Float16* Wt1 = Wt0 + (size_t)DDIM * DDIM;

  auto cvtA = [&](const float* src_, int count) {
    cvt_f32_f16<<<(count + 255) / 256, 256, 0, stream>>>(src_, Ah, count);
  };
  auto cvtW = [&](const float* Wm, _Float16* Wt, int K, int ncols) {
    cvt_w_t<<<(K * ncols + 255) / 256, 256, 0, stream>>>(Wm, Wt, K, ncols);
  };
  auto gemm = [&](const _Float16* A_, const _Float16* Wt, const float* bias,
                  float* C, int K, int ncols) {
    int waves = (ncols >= 128) ? 8 : (ncols / 16);
    dim3 grid(n / TILE_M, ncols / (16 * waves));
    gemm_f16wmma_tdm<<<grid, 32 * waves, 0, stream>>>(A_, Wt, bias, C, K, ncols);
  };

  const int EH_BLK   = (Etot * HEADS + 255) / 256;
  const int INIT_BLK = (n * DDIM + 255) / 256;

  const float* h = x;   // layer-0 input
  for (int l = 0; l < 3; ++l) {
    const int K = (l == 0) ? 64 : DDIM;
    cvtA(h, n * K);                               // Ah <- f16(h)
    cvtW(Wl[l], Wt0, K, DDIM);
    cvtW(Wr[l], Wt1, K, DDIM);
    gemm(Ah, Wt0, bl[l], xlp, K, DDIM);
    gemm(Ah, Wt1, br[l], xrp, K, DDIM);

    float* accp; const float* idn;
    if (l == 0) {
      cvtW(res_W, Wt0, 64, DDIM);
      gemm(Ah, Wt0, res_b, bufA, 64, DDIM);       // residual projection of x
      idn = bufA; accp = bufB;
    } else if (l == 1) { idn = bufB; accp = bufA; }
    else               { idn = bufA; accp = bufB; }

    init_layer<<<INIT_BLK, 256, 0, stream>>>(accp, smaxb, denb, n);
    edge_score<<<EH_BLK, 256, 0, stream>>>(xlp, xrp, att[l], esrc, edst, E, Etot, sbuf, smaxb);
    edge_exp<<<EH_BLK, 256, 0, stream>>>(sbuf, smaxb, denb, edst, E, Etot);
    edge_scatter<<<EH_BLK, 256, 0, stream>>>(sbuf, denb, xlp, esrc, edst, E, Etot, accp);
    node_post<<<(n + 7) / 8, 256, 0, stream>>>(accp, gb[l], ln_g[l], ln_b[l], idn, n);
    h = accp;
  }

  // final projection 256 -> 64 (reuse xl buffer), pooling, MLP head
  cvtA(h, n * DDIM);
  cvtW(fp_W, Wt0, DDIM, 64);
  gemm(Ah, Wt0, fp_b, xlp, DDIM, 64);
  pool_kernel<<<G, 256, 0, stream>>>(xlp, pool_w, embb, P);
  mlp_head<<<1, 512, 0, stream>>>(embb, strategy, W1, b1, g1, bb1,
                                  W2, b2, g2, bb2, W3, b3, (float*)d_out);
}
